// RNN_16535624089705
// MI455X (gfx1250) — compile-verified, matrix operand order checked by reading, not run on hardware
//
#include <hip/hip_runtime.h>
#include <stdint.h>

typedef __attribute__((ext_vector_type(16))) __bf16 v16bf;
typedef __attribute__((ext_vector_type(8)))  __bf16 v8bf;
typedef __attribute__((ext_vector_type(4)))  __bf16 v4bf;
typedef __attribute__((ext_vector_type(8)))  float  v8f;
typedef __attribute__((ext_vector_type(4)))  float  v4f;
typedef unsigned int __attribute__((ext_vector_type(4))) uint4v;
typedef int          __attribute__((ext_vector_type(8))) int8v;
typedef int          __attribute__((ext_vector_type(4))) int4v;

#define DEV __device__ __forceinline__

static constexpr int B_   = 4;
static constexpr int S_   = 4096;
static constexpr int DIN  = 1024;
static constexpr int DST  = 1024;
static constexpr int DOUT = 1024;
static constexpr int NH   = 16;
static constexpr int HD   = 64;
static constexpr int M_   = B_ * S_;     // 16384 rows in both large GEMMs

// LDS tile geometry: 32 bf16 per row (64B) + TDM pad of 16B -> 80B pitch.
static constexpr int KSTEP     = 32;
static constexpr int LDS_PITCH = 40;     // bf16 elements per padded row
static constexpr int BM        = 128;    // block rows (8 waves x 16)
static constexpr int BN        = 64;     // block cols

// ---- bf16 <-> f32 helpers (RNE) ----
DEV __bf16 f2bf(float f) {
  uint32_t u = __builtin_bit_cast(uint32_t, f);
  u += 0x7FFFu + ((u >> 16) & 1u);
  unsigned short s = (unsigned short)(u >> 16);
  return __builtin_bit_cast(__bf16, s);
}
DEV float bf2f(__bf16 b) {
  uint32_t u = ((uint32_t)__builtin_bit_cast(unsigned short, b)) << 16;
  return __builtin_bit_cast(float, u);
}

// Hardware tanh (CDNA5 V_TANH_F32); v_nop covers the TRANS-op RAW hazard.
DEV float fast_tanh(float x) {
  float y;
  asm volatile("v_tanh_f32 %0, %1\n\tv_nop" : "=v"(y) : "v"(x));
  return y;
}

// ---------------------------------------------------------------------------
// TDM: 2-D tile load global -> LDS. D# bit layout per CDNA5 ISA 08 §8.3/8.4.
// data_size = 2 bytes; LDS padding: 4 DWORDs after every 16 DWORDs (80B pitch).
// Toolchain uses the 6-arg builtin: (g0, g1, g2, g3, g4, cpol).
// ---------------------------------------------------------------------------
DEV void tdm_load_2d(uint32_t lds_off, const __bf16* gsrc,
                     uint32_t tile_d0, uint32_t tile_d1, uint32_t stride_elems) {
  const uint64_t ga = (uint64_t)(uintptr_t)gsrc;
  uint4v g0;
  g0[0] = 1u;                                        // count=1, user descriptor
  g0[1] = lds_off;                                   // lds_addr (bytes)
  g0[2] = (uint32_t)ga;                              // global_addr[31:0]
  g0[3] = (uint32_t)((ga >> 32) & 0x01FFFFFFu) | 0x80000000u; // [56:32] | type=2

  const uint32_t td0 = 0x40000000u;                  // huge tensor dims: no OOB
  const uint32_t td1 = 0x40000000u;
  int8v g1;
  g1[0] = (int)((1u << 16) |                         // data_size = 2 bytes
                (1u << 20) |                         // pad_enable
                (3u << 22) |                         // pad_interval: 16 DWORDs
                (3u << 25));                         // pad_amount:   4 DWORDs
  g1[1] = (int)((td0 & 0xFFFFu) << 16);              // tensor_dim0[15:0]
  g1[2] = (int)(((td0 >> 16) & 0xFFFFu) | ((td1 & 0xFFFFu) << 16));
  g1[3] = (int)(((td1 >> 16) & 0xFFFFu) | (tile_d0 << 16)); // tile_dim0
  g1[4] = (int)(tile_d1 & 0xFFFFu);                  // tile_dim1 (tile_dim2=0)
  g1[5] = (int)stride_elems;                         // tensor_dim0_stride[31:0]
  g1[6] = 0;                                         // stride[47:32]=0, dim1_stride=0
  g1[7] = 0;
  const int4v gz4 = {0, 0, 0, 0};                    // 2-D: groups 2/3 unused
  const int8v gz8 = {0, 0, 0, 0, 0, 0, 0, 0};
  __builtin_amdgcn_tensor_load_to_lds(g0, g1, gz4, gz4, gz8, 0);
}

// ---------------------------------------------------------------------------
// Bulk f32 -> bf16 conversion (memory bound; 4 elems/thread)
// ---------------------------------------------------------------------------
__global__ __launch_bounds__(256)
void cvt_f32_to_bf16(const float* __restrict__ in, __bf16* __restrict__ out,
                     int n4) {
  const int i = blockIdx.x * 256 + threadIdx.x;
  if (i < n4) {
    v4f v = ((const v4f*)in)[i];
    v4bf r;
#pragma unroll
    for (int j = 0; j < 4; ++j) r[j] = f2bf(v[j]);
    ((v4bf*)out)[i] = r;
  }
}

// WMMA fragment from a padded LDS row (lane layout per ISA §7.12.2):
// lane L holds row (L&15); half = L>>4 selects K chunks {8h..8h+7, 16+8h..}.
DEV v16bf lds_frag(const __bf16* __restrict__ row, int half) {
  v8bf lo = *(const v8bf*)(row + 8 * half);
  v8bf hi = *(const v8bf*)(row + 16 + 8 * half);
  v16bf r;
#pragma unroll
  for (int i = 0; i < 8; ++i) { r[i] = lo[i]; r[8 + i] = hi[i]; }
  return r;
}

// ---------------------------------------------------------------------------
// GEMM (NT), bf16, TDM double-buffered through LDS.
// Block: 256 threads = 8 waves; block tile 128x64; wave tile 16x64.
// Wave 0 drives the TDM pipeline; compute overlaps the next tile's DMA.
// ---------------------------------------------------------------------------
template <bool OUT_BF16, bool BIAS>
__global__ __launch_bounds__(256)
void gemm_nt_bf16(const __bf16* __restrict__ A, const __bf16* __restrict__ W,
                  const float* __restrict__ bias, void* __restrict__ Out,
                  int M, int N, int K) {
  const int lane    = threadIdx.x & 31;
  const int wave    = threadIdx.x >> 5;
  const int half    = lane >> 4;
  const int mrow    = lane & 15;
  const int rowBase = blockIdx.x * BM + wave * 16;
  const int colBase = blockIdx.y * BN;

  // Double-buffered padded tiles: per buffer A 128x40 + B 64x40 bf16 = 15KB.
  __shared__ __align__(16) __bf16 smem[2][(BM + BN) * LDS_PITCH];

  const __bf16* gA = A + (size_t)blockIdx.x * BM * K;   // block A rows
  const __bf16* gW = W + (size_t)colBase * K;           // block B rows

  v8f acc[4] = {{}, {}, {}, {}};

  // Prologue: DMA tiles for kb=0 into buffer 0.
  if (wave == 0) {
    tdm_load_2d((uint32_t)(uintptr_t)&smem[0][0], gA, KSTEP, BM, (uint32_t)K);
    tdm_load_2d((uint32_t)(uintptr_t)&smem[0][BM * LDS_PITCH], gW, KSTEP, BN,
                (uint32_t)K);
  }

  int cur = 0;
  for (int kb = 0; kb < K; kb += KSTEP) {
    const int nxt = kb + KSTEP;
    if (wave == 0) {
      if (nxt < K) {
        tdm_load_2d((uint32_t)(uintptr_t)&smem[cur ^ 1][0], gA + nxt, KSTEP, BM,
                    (uint32_t)K);
        tdm_load_2d((uint32_t)(uintptr_t)&smem[cur ^ 1][BM * LDS_PITCH],
                    gW + nxt, KSTEP, BN, (uint32_t)K);
        __builtin_amdgcn_s_wait_tensorcnt(2);  // current buffer complete
      } else {
        __builtin_amdgcn_s_wait_tensorcnt(0);
      }
    }
    __syncthreads();  // current tiles visible to all waves

    const __bf16* At = &smem[cur][0];
    const __bf16* Bt = &smem[cur][BM * LDS_PITCH];
    v16bf a = lds_frag(At + (wave * 16 + mrow) * LDS_PITCH, half);
#pragma unroll
    for (int nt = 0; nt < 4; ++nt) {
      v16bf b = lds_frag(Bt + (nt * 16 + mrow) * LDS_PITCH, half);
      acc[nt] = __builtin_amdgcn_wmma_f32_16x16x32_bf16(
          false, a, false, b, (short)0, acc[nt], false, false);
    }

    __syncthreads();  // reads done before wave0 DMAs into this buffer again
    cur ^= 1;
  }

#pragma unroll
  for (int nt = 0; nt < 4; ++nt) {
    const int n  = colBase + nt * 16 + mrow;
    const float bv = BIAS ? bias[n] : 0.0f;
#pragma unroll
    for (int r = 0; r < 8; ++r) {
      const int m  = rowBase + r + 8 * half;
      const float v = acc[nt][r] + bv;
      if constexpr (OUT_BF16)
        ((__bf16*)Out)[(size_t)m * N + n] = f2bf(v);
      else
        ((float*)Out)[(size_t)m * N + n] = v;
    }
  }
}

// ---------------------------------------------------------------------------
// Sequential scan: one block per head, 4 waves; wave w owns h_new columns
// [16w,16w+16). h kept in LDS as bf16 padded to 16 rows (rows 4..15 zero).
// ---------------------------------------------------------------------------
__global__ __launch_bounds__(128)
void rnn_scan(const __bf16* __restrict__ proj,     // (B,S,1024) bf16
              const float*  __restrict__ h0,       // (B,1024) f32
              const float*  __restrict__ Wst,      // (NH,64,64) f32: [h][k_out]
              __bf16*       __restrict__ states,   // (B,S,1024) bf16
              float*        __restrict__ finalOut) // (B,1024) f32
{
  const int head = blockIdx.x;
  const int tid  = threadIdx.x;
  const int lane = tid & 31;
  const int wave = tid >> 5;          // n-tile index 0..3
  const int half = lane >> 4;
  const int mrow = lane & 15;

  __shared__ __align__(16) __bf16 hbuf[16][HD];  // rows >= B_ stay zero

  for (int i = tid; i < 16 * HD; i += 128) {
    const int r = i / HD, k = i % HD;
    hbuf[r][k] = f2bf(r < B_ ? h0[r * DST + head * HD + k] : 0.0f);
  }

  // Preload W_head fragments (stride-HD gather, one-time). B[k_h][n]=W[k_h*HD+n].
  const float* W = Wst + (size_t)head * HD * HD;
  const int n = wave * 16 + mrow;
  v16bf bfrag[2];
#pragma unroll
  for (int kc = 0; kc < 2; ++kc) {
    const int k0 = kc * 32 + 8 * half;
    const int k1 = kc * 32 + 16 + 8 * half;
    v16bf f;
#pragma unroll
    for (int i = 0; i < 8; ++i) {
      f[i]     = f2bf(W[(k0 + i) * HD + n]);
      f[8 + i] = f2bf(W[(k1 + i) * HD + n]);
    }
    bfrag[kc] = f;
  }
  __syncthreads();

  const size_t colOff = (size_t)head * HD + wave * 16 + mrow;

  for (int t = 0; t < S_; ++t) {
    v16bf a0, a1;
    {
      const __bf16* hp = &hbuf[mrow][0];
      v8bf x0 = *(const v8bf*)(hp + 8 * half);
      v8bf x1 = *(const v8bf*)(hp + 16 + 8 * half);
      v8bf x2 = *(const v8bf*)(hp + 32 + 8 * half);
      v8bf x3 = *(const v8bf*)(hp + 48 + 8 * half);
#pragma unroll
      for (int i = 0; i < 8; ++i) {
        a0[i] = x0[i]; a0[8 + i] = x1[i];
        a1[i] = x2[i]; a1[8 + i] = x3[i];
      }
    }

    v8f acc = {};
    if (half == 0) {
#pragma unroll
      for (int r = 0; r < 4; ++r)
        acc[r] = bf2f(proj[((size_t)r * S_ + t) * DST + colOff]);
    }

    acc = __builtin_amdgcn_wmma_f32_16x16x32_bf16(
        false, a0, false, bfrag[0], (short)0, acc, false, false);
    acc = __builtin_amdgcn_wmma_f32_16x16x32_bf16(
        false, a1, false, bfrag[1], (short)0, acc, false, false);

    __syncthreads();  // all waves finished reading hbuf(t)
    if (half == 0) {
#pragma unroll
      for (int r = 0; r < 4; ++r) {
        const __bf16 bv = f2bf(fast_tanh(acc[r]));
        hbuf[r][wave * 16 + mrow] = bv;
        states[((size_t)r * S_ + t) * DST + colOff] = bv;
      }
    }
    __syncthreads();  // hbuf(t+1) visible
  }

  for (int i = tid; i < B_ * HD; i += 128) {
    const int b = i / HD, k = i % HD;
    finalOut[b * DST + head * HD + k] = bf2f(hbuf[b][k]);
  }
}

// ---------------------------------------------------------------------------
extern "C" void kernel_launch(void* const* d_in, const int* in_sizes, int n_in,
                              void* d_out, int out_size, void* d_ws, size_t ws_size,
                              hipStream_t stream) {
  const float* x    = (const float*)d_in[0];  // (B,S,DIN)
  const float* h0   = (const float*)d_in[1];  // (B,DST)
  const float* w_in = (const float*)d_in[2];  // (DST,DIN)
  const float* b_in = (const float*)d_in[3];  // (DST)
  const float* wst  = (const float*)d_in[4];  // (NH,HD,HD)
  const float* wout = (const float*)d_in[5];  // (DOUT,DST)

  float* out      = (float*)d_out;            // (B,S,DOUT)
  float* finalOut = out + (size_t)M_ * DOUT;  // (B,DST)

  // Workspace layout (bf16 elements)
  __bf16* xbf    = (__bf16*)d_ws;                       // M_*DIN
  __bf16* winbf  = xbf   + (size_t)M_ * DIN;            // DST*DIN
  __bf16* woutbf = winbf + (size_t)DST * DIN;           // DOUT*DST
  __bf16* proj   = woutbf + (size_t)DOUT * DST;         // M_*DST
  __bf16* states = proj  + (size_t)M_ * DST;            // M_*DST

  // 0) one-time f32 -> bf16 conversions (memory bound)
  {
    const int n4x = (M_ * DIN) / 4;
    cvt_f32_to_bf16<<<(n4x + 255) / 256, 256, 0, stream>>>(x, xbf, n4x);
    const int n4w = (DST * DIN) / 4;
    cvt_f32_to_bf16<<<(n4w + 255) / 256, 256, 0, stream>>>(w_in, winbf, n4w);
    const int n4o = (DOUT * DST) / 4;
    cvt_f32_to_bf16<<<(n4o + 255) / 256, 256, 0, stream>>>(wout, woutbf, n4o);
  }

  dim3 blk(256);

  // 1) proj = x @ w_in^T + b_in   (bf16 in, bf16 out, TDM pipeline)
  gemm_nt_bf16<true, true><<<dim3(M_ / BM, DST / BN), blk, 0, stream>>>(
      xbf, winbf, b_in, (void*)proj, M_, DST, DIN);

  // 2) sequential tanh recurrence over S
  rnn_scan<<<NH, 128, 0, stream>>>(proj, h0, wst, states, finalOut);

  // 3) out = states @ w_out^T     (bf16 in, f32 out, TDM pipeline)
  gemm_nt_bf16<false, false><<<dim3(M_ / BM, DOUT / BN), blk, 0, stream>>>(
      states, woutbf, nullptr, (void*)out, M_, DOUT, DST);
}
